// Beit3MultiheadAttention_41575283425293
// MI455X (gfx1250) — compile-verified, matrix operand order checked by reading, not run on hardware
//
#include <hip/hip_runtime.h>
#include <hip/hip_bf16.h>
#include <math.h>

// BEiT3 multihead attention, MI455X (gfx1250), bf16 WMMA pipeline.
#define EMB 1024
#define SEQ 1024
#define NH  16
#define HD  64
#define BATCH 4

typedef __bf16 bfx8  __attribute__((ext_vector_type(8)));
typedef __bf16 bfx16 __attribute__((ext_vector_type(16)));
typedef float  fx8   __attribute__((ext_vector_type(8)));

union FragBF { bfx16 v; bfx8 h[2]; };

// round-to-nearest-even fp32 -> bf16
__device__ __forceinline__ __bf16 f2bf(float f) {
    unsigned u = __builtin_bit_cast(unsigned, f);
    u += 0x7FFFu + ((u >> 16) & 1u);
    unsigned short s = (unsigned short)(u >> 16);
    return __builtin_bit_cast(__bf16, s);
}

// 16-bit A/B fragment (16x32 / 32x16) from a row-major bf16 matrix, leading dim ld.
// Lane (l&15) -> M (or N) index; lanes 0-15 carry K {0..7,16..23}, lanes 16-31 K {8..15,24..31}.
__device__ __forceinline__ FragBF load_frag_bf16(const __bf16* tile, int ld) {
    const int lane = threadIdx.x & 31;
    const __bf16* p = tile + (size_t)(lane & 15) * ld + ((lane >> 4) << 3);
    FragBF f;
    f.h[0] = *(const bfx8*)(p);
    f.h[1] = *(const bfx8*)(p + 16);
    return f;
}

__device__ __forceinline__ fx8 wmma_bf16(const FragBF& a, const FragBF& b, fx8 c) {
    return __builtin_amdgcn_wmma_f32_16x16x32_bf16(false, a.v, false, b.v,
                                                   (short)0, c, false, false);
}

// ---------------------------------------------------------------- fp32 -> bf16 convert
__global__ void conv_bf16_kernel(const float* __restrict__ src, __bf16* __restrict__ dst, int n) {
    for (int i = blockIdx.x * blockDim.x + threadIdx.x; i < n; i += gridDim.x * blockDim.x)
        dst[i] = f2bf(src[i]);
}

// ---------------------------------------------------------------- multiway projection GEMM
// out[row, o] = (sum_e X[row, e] * W[wi, o, e] + bias[wi, o]) * scale
// wi = ((row % SEQ) < sp) ? 0 : 1  (sp is a multiple of 16 -> uniform per 16-row tile).
// mode 0: store bf16 [rows, EMB]            (Q, K)
// mode 1: store bf16 transposed [B,H,D,S]   (V, so P@V has contiguous-K B-fragments)
// mode 2: store fp32 [rows, EMB]            (final output projection)
__device__ __forceinline__ void store_tile(fx8 acc, int mbase, int nbase, int wi,
                                           const float* __restrict__ bias2, float scale,
                                           int mode, float* __restrict__ outF,
                                           __bf16* __restrict__ outB) {
    const int lane = threadIdx.x & 31;
    const int col  = nbase + (lane & 15);
    const float bias = bias2[wi * EMB + col];
#pragma unroll
    for (int r = 0; r < 8; ++r) {
        int row = mbase + r + ((lane >> 4) << 3);
        float v = (acc[r] + bias) * scale;
        if (mode == 2) {
            outF[(size_t)row * EMB + col] = v;
        } else if (mode == 0) {
            outB[(size_t)row * EMB + col] = f2bf(v);
        } else {
            int bb = row >> 10, s = row & (SEQ - 1);
            int hh = col >> 6,  dd = col & (HD - 1);
            outB[((size_t)(bb * NH + hh) * HD + dd) * SEQ + s] = f2bf(v);
        }
    }
}

// Wave tile: 32 rows x 64 cols (2 M-subtiles x 4 N-subtiles = 8 WMMA accumulators).
// Block: 4 waves in a 2x2 arrangement -> 64 x 128 block tile.
__global__ void proj_kernel(const __bf16* __restrict__ X, const __bf16* __restrict__ W2,
                            const float* __restrict__ bias2, const int* __restrict__ spp,
                            float scale, int mode, float* __restrict__ outF,
                            __bf16* __restrict__ outB) {
    const int sp   = *spp;
    const int wave = threadIdx.x >> 5;
    const int m0   = blockIdx.x * 64 + (wave >> 1) * 32;
    const int n0   = blockIdx.y * 128 + (wave & 1) * 64;
    const int wi0  = (((m0)      & (SEQ - 1)) < sp) ? 0 : 1;
    const int wi1  = (((m0 + 16) & (SEQ - 1)) < sp) ? 0 : 1;
    const __bf16* xb = X + (size_t)m0 * EMB;

    fx8 acc[2][4] = {};
    if (wi0 == wi1) {
        // common case: both 16-row subtiles use the same weight matrix ->
        // 12 b128 loads per 8 WMMAs, full B-fragment reuse across M.
        const __bf16* wb = W2 + (size_t)wi0 * EMB * EMB + (size_t)n0 * EMB;
        for (int k = 0; k < EMB; k += 32) {
            FragBF a0 = load_frag_bf16(xb + k, EMB);
            FragBF a1 = load_frag_bf16(xb + (size_t)16 * EMB + k, EMB);
            __builtin_prefetch(wb + k + 64, 0, 1);            // global_prefetch_b8
            FragBF b[4];
#pragma unroll
            for (int j = 0; j < 4; ++j)
                b[j] = load_frag_bf16(wb + (size_t)(j * 16) * EMB + k, EMB);
#pragma unroll
            for (int j = 0; j < 4; ++j) acc[0][j] = wmma_bf16(a0, b[j], acc[0][j]);
#pragma unroll
            for (int j = 0; j < 4; ++j) acc[1][j] = wmma_bf16(a1, b[j], acc[1][j]);
        }
    } else {
        // straddle case (at most one 16-row band per sequence): separate B fragments.
        const __bf16* wbl = W2 + (size_t)wi0 * EMB * EMB + (size_t)n0 * EMB;
        const __bf16* wbh = W2 + (size_t)wi1 * EMB * EMB + (size_t)n0 * EMB;
        for (int k = 0; k < EMB; k += 32) {
            FragBF a0 = load_frag_bf16(xb + k, EMB);
            FragBF a1 = load_frag_bf16(xb + (size_t)16 * EMB + k, EMB);
            FragBF bl[4], bh[4];
#pragma unroll
            for (int j = 0; j < 4; ++j) {
                bl[j] = load_frag_bf16(wbl + (size_t)(j * 16) * EMB + k, EMB);
                bh[j] = load_frag_bf16(wbh + (size_t)(j * 16) * EMB + k, EMB);
            }
#pragma unroll
            for (int j = 0; j < 4; ++j) acc[0][j] = wmma_bf16(a0, bl[j], acc[0][j]);
#pragma unroll
            for (int j = 0; j < 4; ++j) acc[1][j] = wmma_bf16(a1, bh[j], acc[1][j]);
        }
    }
#pragma unroll
    for (int j = 0; j < 4; ++j) {
        store_tile(acc[0][j], m0,      n0 + j * 16, wi0, bias2, scale, mode, outF, outB);
        store_tile(acc[1][j], m0 + 16, n0 + j * 16, wi1, bias2, scale, mode, outF, outB);
    }
}

// ---------------------------------------------------------------- attention core
// One block: (b,h) pair, 32 query rows. 8 waves, 256 threads.
// LDS: 32x1024 fp32 scores (128 KB) + 32x1024 bf16 probs (64 KB) = 192 KB dynamic.
__global__ void attn_kernel(const __bf16* __restrict__ qb, const __bf16* __restrict__ kb,
                            const __bf16* __restrict__ vT, const float* __restrict__ amask,
                            const unsigned char* __restrict__ kpm,
                            const float* __restrict__ rel,
                            float* __restrict__ attnw_out,   // [H,B,T,S]
                            float* __restrict__ attn_out) {  // [B,T,E]
    extern __shared__ float sS[];                            // [32][1024] fp32
    __bf16* sP = (__bf16*)(sS + 32 * SEQ);                   // [32][1024] bf16
    const int lane = threadIdx.x & 31;
    const int wave = threadIdx.x >> 5;                       // 0..7
    const int bh = blockIdx.y, b = bh >> 4, h = bh & (NH - 1);
    const int t0 = blockIdx.x * 32;

    // ---- phase 1: S = Q K^T (+ masks + relative position) ----
    const __bf16* Qb = qb + ((size_t)(b * SEQ + t0)) * EMB + h * HD;
    FragBF a00 = load_frag_bf16(Qb, EMB);
    FragBF a01 = load_frag_bf16(Qb + 32, EMB);
    FragBF a10 = load_frag_bf16(Qb + (size_t)16 * EMB, EMB);
    FragBF a11 = load_frag_bf16(Qb + (size_t)16 * EMB + 32, EMB);

    for (int nt = 0; nt < 8; ++nt) {
        const int s0 = wave * 128 + nt * 16;
        const __bf16* Kb = kb + ((size_t)(b * SEQ + s0)) * EMB + h * HD;
        FragBF bk0 = load_frag_bf16(Kb, EMB);
        FragBF bk1 = load_frag_bf16(Kb + 32, EMB);
        fx8 acc0 = {}, acc1 = {};
        acc0 = wmma_bf16(a00, bk0, acc0);
        acc0 = wmma_bf16(a01, bk1, acc0);
        acc1 = wmma_bf16(a10, bk0, acc1);
        acc1 = wmma_bf16(a11, bk1, acc1);

        const int s = s0 + (lane & 15);
        const bool pad = kpm[b * SEQ + s] != 0;
#pragma unroll
        for (int r = 0; r < 8; ++r) {
            int tt = r + ((lane >> 4) << 3);                 // acc0 rows 0..15
            int t  = t0 + tt;
            float v = acc0[r] + amask[(size_t)t * SEQ + s]
                    + rel[(((size_t)bh) * SEQ + t) * SEQ + s];
            sS[tt * SEQ + s] = pad ? -INFINITY : v;
            tt += 16;  t = t0 + tt;                          // acc1 rows 16..31
            v = acc1[r] + amask[(size_t)t * SEQ + s]
              + rel[(((size_t)bh) * SEQ + t) * SEQ + s];
            sS[tt * SEQ + s] = pad ? -INFINITY : v;
        }
    }
    __syncthreads();

    // ---- phase 2: fp32 softmax; emit attn_weights [H,B,T,S] + bf16 probs to LDS ----
    for (int rr = 0; rr < 4; ++rr) {
        const int tr = wave * 4 + rr;
        float* row = sS + tr * SEQ;
        float m = -INFINITY;
        for (int i = lane; i < SEQ; i += 32) m = fmaxf(m, row[i]);
#pragma unroll
        for (int off = 16; off; off >>= 1) m = fmaxf(m, __shfl_xor(m, off, 32));
        float sum = 0.f;
        for (int i = lane; i < SEQ; i += 32) {
            float e = __expf(row[i] - m);
            row[i] = e;
            sum += e;
        }
#pragma unroll
        for (int off = 16; off; off >>= 1) sum += __shfl_xor(sum, off, 32);
        const float inv = 1.0f / sum;
        const size_t obase = (((size_t)(h * BATCH + b)) * SEQ + (t0 + tr)) * SEQ;
        for (int i = lane; i < SEQ; i += 32) {
            float p = row[i] * inv;
            attnw_out[obase + i] = p;
            sP[tr * SEQ + i] = f2bf(p);
        }
    }
    __syncthreads();

    // ---- phase 3: O = P V  (bf16 A from LDS, bf16 B from transposed V) ----
    const int mt = wave >> 2, nt = wave & 3;                 // 2 x 4 tiles, one per wave
    fx8 acc = {};
    const __bf16* Vb = vT + ((size_t)(b * NH + h)) * HD * SEQ + (size_t)(nt * 16) * SEQ;
    const __bf16* Pb = sP + (size_t)mt * 16 * SEQ;
    for (int k = 0; k < SEQ; k += 32) {
        FragBF a  = load_frag_bf16(Pb + k, SEQ);
        FragBF bb = load_frag_bf16(Vb + k, SEQ);
        acc = wmma_bf16(a, bb, acc);
    }
    const int d = h * HD + nt * 16 + (lane & 15);
#pragma unroll
    for (int r = 0; r < 8; ++r) {
        int t = t0 + mt * 16 + r + ((lane >> 4) << 3);
        attn_out[((size_t)(b * SEQ + t)) * EMB + d] = acc[r];
    }
}

// ---------------------------------------------------------------- multiway layernorm -> bf16
__global__ void ln_kernel(const float* __restrict__ X, const float* __restrict__ g2,
                          const float* __restrict__ b2, const int* __restrict__ spp,
                          __bf16* __restrict__ out) {
    const int row  = blockIdx.x * 8 + (threadIdx.x >> 5);    // one wave per token
    const int lane = threadIdx.x & 31;
    const int wi   = ((row & (SEQ - 1)) < *spp) ? 0 : 1;
    const float* x = X + (size_t)row * EMB;
    float s = 0.f;
    for (int i = lane; i < EMB; i += 32) s += x[i];
#pragma unroll
    for (int off = 16; off; off >>= 1) s += __shfl_xor(s, off, 32);
    const float mean = s * (1.0f / EMB);
    float v = 0.f;
    for (int i = lane; i < EMB; i += 32) { float d = x[i] - mean; v += d * d; }
#pragma unroll
    for (int off = 16; off; off >>= 1) v += __shfl_xor(v, off, 32);
    const float rstd = rsqrtf(v * (1.0f / EMB) + 1e-5f);
    __bf16* o = out + (size_t)row * EMB;
    for (int i = lane; i < EMB; i += 32)
        o[i] = f2bf((x[i] - mean) * rstd * g2[wi * EMB + i] + b2[wi * EMB + i]);
}

// ---------------------------------------------------------------- host launch
extern "C" void kernel_launch(void* const* d_in, const int* in_sizes, int n_in,
                              void* d_out, int out_size, void* d_ws, size_t ws_size,
                              hipStream_t stream) {
    const float* q   = (const float*)d_in[0];
    const float* k   = (const float*)d_in[1];
    const float* v   = (const float*)d_in[2];
    const unsigned char* kpm = (const unsigned char*)d_in[3];
    const float* amask = (const float*)d_in[4];
    const float* rel   = (const float*)d_in[5];
    const float* qw = (const float*)d_in[6];   const float* qbias = (const float*)d_in[7];
    const float* kw = (const float*)d_in[8];   const float* kbias = (const float*)d_in[9];
    const float* vw = (const float*)d_in[10];  const float* vbias = (const float*)d_in[11];
    const float* ow = (const float*)d_in[12];  const float* obias = (const float*)d_in[13];
    const float* ln_g = (const float*)d_in[14];
    const float* ln_b = (const float*)d_in[15];
    const int*   spp  = (const int*)d_in[16];

    // workspace carve-up (with lifetime aliasing):
    //   [ 0,16) MB : bf16 weights wq|wk|wv|wo
    //   [16,40) MB : qbf | kbf | vT (bf16 activations, live until end of attention)
    //   [40,64) MB : phase A: xq|xk|xv bf16 input staging (dead after projections)
    //                phase B: aout fp32 [40,56) + lnbf bf16 [56,64)
    char* w = (char*)d_ws;
    const size_t MB = 1024 * 1024;
    __bf16* wq  = (__bf16*)(w);
    __bf16* wk  = (__bf16*)(w + 4 * MB);
    __bf16* wv  = (__bf16*)(w + 8 * MB);
    __bf16* wo  = (__bf16*)(w + 12 * MB);
    __bf16* qbf = (__bf16*)(w + 16 * MB);
    __bf16* kbf = (__bf16*)(w + 24 * MB);
    __bf16* vTb = (__bf16*)(w + 32 * MB);
    __bf16* xq  = (__bf16*)(w + 40 * MB);
    __bf16* xk  = (__bf16*)(w + 48 * MB);
    __bf16* xv  = (__bf16*)(w + 56 * MB);
    float*  aout = (float*)(w + 40 * MB);     // aliases xq/xk (phase B)
    __bf16* lnbf = (__bf16*)(w + 56 * MB);    // aliases xv    (phase B)

    const size_t ASZ = (size_t)BATCH * SEQ * EMB;     // 4M elements
    float* outF  = (float*)d_out;                     // [B,T,E]
    float* attnW = outF + ASZ;                        // [H,B,T,S]

    // 1) weights and activations -> bf16
    const int nW = 2 * EMB * EMB;
    conv_bf16_kernel<<<2048, 256, 0, stream>>>(qw, wq, nW);
    conv_bf16_kernel<<<2048, 256, 0, stream>>>(kw, wk, nW);
    conv_bf16_kernel<<<2048, 256, 0, stream>>>(vw, wv, nW);
    conv_bf16_kernel<<<2048, 256, 0, stream>>>(ow, wo, nW);
    conv_bf16_kernel<<<4096, 256, 0, stream>>>(q, xq, (int)ASZ);
    conv_bf16_kernel<<<4096, 256, 0, stream>>>(k, xk, (int)ASZ);
    conv_bf16_kernel<<<4096, 256, 0, stream>>>(v, xv, (int)ASZ);

    // 2) Q/K/V projections (rows = B*S = 4096; block tile 64x128)
    const dim3 pg(BATCH * SEQ / 64, EMB / 128);
    const float scaling = 0.125f;                     // HD^-0.5
    proj_kernel<<<pg, 128, 0, stream>>>(xq, wq, qbias, spp, scaling, 0, nullptr, qbf);
    proj_kernel<<<pg, 128, 0, stream>>>(xk, wk, kbias, spp, 1.0f,    0, nullptr, kbf);
    proj_kernel<<<pg, 128, 0, stream>>>(xv, wv, vbias, spp, 1.0f,    1, nullptr, vTb);

    // 3) attention (192 KB dynamic LDS; WGP has 320 KB)
    const int smem = 32 * SEQ * (int)sizeof(float) + 32 * SEQ * (int)sizeof(__bf16);
    hipFuncSetAttribute((const void*)attn_kernel,
                        hipFuncAttributeMaxDynamicSharedMemorySize, smem);
    attn_kernel<<<dim3(SEQ / 32, BATCH * NH), 256, smem, stream>>>(
        qbf, kbf, vTb, amask, kpm, rel, attnW, aout);

    // 4) inner layernorm -> bf16
    ln_kernel<<<BATCH * SEQ / 8, 256, 0, stream>>>(aout, ln_g, ln_b, spp, lnbf);

    // 5) output projection -> d_out
    proj_kernel<<<pg, 128, 0, stream>>>(lnbf, wo, obias, spp, 1.0f, 2, outF, nullptr);
}
